// GaussianQuantRegularizer_867583393938
// MI455X (gfx1250) — compile-verified, matrix operand order checked by reading, not run on hardware
//
#include <hip/hip_runtime.h>

// GaussianQuantRegularizer for MI455X (gfx1250, wave32).
//
// argmax_n of the score reduces (dropping row-constant / n-constant terms) to
// a fused 65536x16 @ 16x2048 bf16 GEMM + row-argmax on v_wmma_f32_16x16x32_bf16.
// Each wave owns TWO 16-row tiles so every candidate B fragment feeds 2 WMMAs
// (halves VMEM per unit of matrix work); B fetches are software-pipelined.

typedef __attribute__((ext_vector_type(16))) __bf16 v16bf;
typedef __attribute__((ext_vector_type(8)))  float  v8f;

#define NCAND   2048          // prior samples
#define GROUPSZ 8
#define NG      4             // c2 / GROUP
#define C2      32
#define CCH     64            // channels in z
#define LPIX    4096          // h*w
#define BATCH   4
#define NROWS   (BATCH * LPIX * NG)   // 65536
#define NTILE   (NCAND / 16)          // 128 candidate tiles
#define RTILES  (NROWS / 16)          // 4096 row tiles
#define MT      2                     // row tiles per wave

// ---------------------------------------------------------------------------
// Kernel 1: build candidate feature matrix in WMMA B-fragment layout.
// Features per candidate n (K dim, padded 16 -> 32):
//   K 0..7 : s[n,g]^2    K 8..15 : s[n,g]    K 16..31 : 0 (pad)
// B-matrix (32x16 bf16) lane layout: lanes 0-15 hold N=lane, K {0..7,16..23};
// lanes 16-31 hold N=lane-16, K {8..15,24..31}. Stored as [tile][lane][16 bf16]
// so the GEMM kernel loads one contiguous v16bf per lane per tile.
// ---------------------------------------------------------------------------
__global__ void gq_build_feat(const float* __restrict__ s,
                              __bf16* __restrict__ Bbuf) {
  int n = blockIdx.x * blockDim.x + threadIdx.x;
  if (n >= NCAND) return;
  const int t = n >> 4, col = n & 15;
  __bf16* p0 = Bbuf + ((size_t)(t * 32 + col)      << 4); // lanes 0-15 slot
  __bf16* p1 = Bbuf + ((size_t)(t * 32 + col + 16) << 4); // lanes 16-31 slot
#pragma unroll
  for (int g = 0; g < 8; ++g) {
    float v = s[(size_t)n * GROUPSZ + g];
    p0[g]     = (__bf16)(v * v);  // K = g       (s^2 feature)
    p0[g + 8] = (__bf16)0.0f;     // K = 16 + g  (pad)
    p1[g]     = (__bf16)v;        // K = 8 + g   (s feature)
    p1[g + 8] = (__bf16)0.0f;     // K = 24 + g  (pad)
  }
}

// Build the A fragment (16x32 bf16) for row tile rt, for this lane.
// Lanes 0-15 carry K0..7 = (0.5 - 0.5*inv_var) (multiplies s^2; absorbs the
// -beta*nlp term), lanes 16-31 carry K8..15 = mu*inv_var (multiplies s).
// K16..31 are zero padding.
__device__ __forceinline__ v16bf gq_make_afrag(const float* __restrict__ z,
                                               int rt, int lane) {
  const int m  = lane & 15;
  const int r  = rt * 16 + m;            // logical row (b*L + l)*NG + j
  const int j  = r & (NG - 1);
  const int bl = r >> 2;
  const int l  = bl & (LPIX - 1);
  const int b  = bl >> 12;
  v16bf a;
#pragma unroll
  for (int g = 0; g < 8; ++g) {
    const int ch = g * NG + j;
    float mu = z[((size_t)(b * CCH + ch))      * LPIX + l];
    float lv = z[((size_t)(b * CCH + ch + C2)) * LPIX + l];
    lv = fminf(fmaxf(lv, -30.0f), 20.0f);
    float iv = __expf(-lv);                      // 1/std^2
    float coef = (lane < 16) ? (0.5f - 0.5f * iv) : (mu * iv);
    a[g]     = (__bf16)coef;
    a[g + 8] = (__bf16)0.0f;
  }
  return a;
}

// ---------------------------------------------------------------------------
// Kernel 2: per wave, MT 16-row tiles sharing each B fragment. Sweep all 128
// candidate tiles (software-pipelined B fetch), one WMMA per (row tile x cand
// tile), fuse running argmax, butterfly-reduce across 16-lane halves, scatter.
// ---------------------------------------------------------------------------
__global__ void gq_score_argmax(const float* __restrict__ z,
                                const float* __restrict__ prior,
                                const __bf16* __restrict__ Bbuf,
                                float* __restrict__ zhat,
                                int*   __restrict__ idxo) {
  const int lane = threadIdx.x & 31;
  const int wave = threadIdx.x >> 5;
  const int w    = blockIdx.x * (blockDim.x >> 5) + wave;  // wave id
  const int rt0  = w * MT;                                 // first row tile
  const int m    = lane & 15;

  v16bf a[MT];
#pragma unroll
  for (int u = 0; u < MT; ++u) a[u] = gq_make_afrag(z, rt0 + u, lane);

  float best[MT][8];
  int   bidx[MT][8];
#pragma unroll
  for (int u = 0; u < MT; ++u)
#pragma unroll
    for (int i = 0; i < 8; ++i) { best[u][i] = -3.4e38f; bidx[u][i] = 0; }

  const v16bf* Bt = (const v16bf*)Bbuf;          // [tile*32 + lane]
  v16bf bcur = Bt[lane];                         // prefetch tile 0
#pragma unroll 2
  for (int t = 0; t < NTILE; ++t) {
    // software pipeline: fetch t+1 (wrapped, branch-free) before consuming t
    v16bf bnext = Bt[(size_t)(((t + 1) & (NTILE - 1)) * 32 + lane)];
    const int n0 = t * 16 + m;                   // this lane's candidate id
#pragma unroll
    for (int u = 0; u < MT; ++u) {
      v8f c = {};
      c = __builtin_amdgcn_wmma_f32_16x16x32_bf16(
          /*neg_a=*/false, a[u], /*neg_b=*/false, bcur,
          /*c_mod=*/(short)0, c, /*reuse_a=*/false, /*reuse_b=*/false);
#pragma unroll
      for (int i = 0; i < 8; ++i) {              // C layout: VGPR i -> row
        if (c[i] > best[u][i]) { best[u][i] = c[i]; bidx[u][i] = n0; }
      }
    }
    bcur = bnext;
  }

  // Reduce over the 16 lanes of each half (each holds a distinct n residue).
  // Tie-break on smaller index -> matches jnp.argmax first-occurrence.
#pragma unroll
  for (int off = 1; off < 16; off <<= 1) {
#pragma unroll
    for (int u = 0; u < MT; ++u)
#pragma unroll
      for (int i = 0; i < 8; ++i) {
        float ov = __shfl_xor(best[u][i], off, 32);
        int   oi = __shfl_xor(bidx[u][i], off, 32);
        if (ov > best[u][i] || (ov == best[u][i] && oi < bidx[u][i])) {
          best[u][i] = ov; bidx[u][i] = oi;
        }
      }
  }

  // lane 0 owns C rows M=0..7, lane 16 owns M=8..15 (per tile).
  if (m == 0) {
    const int mbase = (lane >> 4) * 8;
#pragma unroll
    for (int u = 0; u < MT; ++u) {
#pragma unroll
      for (int i = 0; i < 8; ++i) {
        const int rr  = (rt0 + u) * 16 + mbase + i;
        const int jj  = rr & (NG - 1);
        const int bl2 = rr >> 2;
        const int ll  = bl2 & (LPIX - 1);
        const int bb  = bl2 >> 12;
        const int idx = bidx[u][i];
        idxo[((size_t)(bb * NG + jj)) * LPIX + ll] = idx;
#pragma unroll
        for (int g = 0; g < 8; ++g) {
          float q = prior[(size_t)idx * GROUPSZ + g];
          zhat[((size_t)(bb * C2 + g * NG + jj)) * LPIX + ll] = q;
        }
      }
    }
  }
}

// ---------------------------------------------------------------------------
extern "C" void kernel_launch(void* const* d_in, const int* in_sizes, int n_in,
                              void* d_out, int out_size, void* d_ws,
                              size_t ws_size, hipStream_t stream) {
  const float* z     = (const float*)d_in[0];   // (4,64,64,64) f32
  const float* prior = (const float*)d_in[1];   // (2048,8) f32

  __bf16* Bbuf = (__bf16*)d_ws;                 // 128 tiles * 32 lanes * 16 bf16 = 128 KB
  float*  zhat = (float*)d_out;                 // (4,32,64,64) f32
  int*    idxo = (int*)((float*)d_out + (size_t)BATCH * C2 * LPIX); // (4,4,64,64) i32

  gq_build_feat<<<(NCAND + 255) / 256, 256, 0, stream>>>(prior, Bbuf);

  // 4096 row tiles / 2 per wave = 2048 waves; 8 waves per 256-thread block.
  gq_score_argmax<<<(RTILES / MT) / 8, 256, 0, stream>>>(z, prior, Bbuf, zhat, idxo);
}
